// CausalMulHeadAttention_50878182588504
// MI455X (gfx1250) — matrix operand order repaired and checked by reading, not compiled, not run on hardware
//
#include <hip/hip_runtime.h>
#include <hip/hip_bf16.h>

typedef __bf16 bf16;
typedef __attribute__((ext_vector_type(8)))  __bf16 v8bf;
typedef __attribute__((ext_vector_type(16))) __bf16 v16bf;
typedef __attribute__((ext_vector_type(8)))  float  v8f;
typedef __attribute__((ext_vector_type(4)))  unsigned int v4u;
typedef __attribute__((ext_vector_type(8)))  int v8i;
typedef __attribute__((ext_vector_type(4)))  int v4i;

#define B_  4
#define S_  2048
#define D_  1024
#define H_  16
#define HD_ 64
#define NT_ (B_ * S_)   // 8192 tokens

#if defined(__HIP_DEVICE_COMPILE__) && __has_builtin(__builtin_amdgcn_tensor_load_to_lds)
#define HAVE_TDM 1
#else
#define HAVE_TDM 0
#endif

static __device__ __forceinline__ v16bf cat8(v8bf lo, v8bf hi) {
    return __builtin_shufflevector(lo, hi, 0,1,2,3,4,5,6,7,8,9,10,11,12,13,14,15);
}

#if HAVE_TDM
// ---------------------------------------------------------------------------
// TDM 2D tile load: global (row-major, 2-byte elems, row stride in elems)
// -> LDS (tile packed row-major). D# packing per CDNA5 ISA.
// Issue from ONE wave only; complete with s_wait_tensorcnt(0) + barrier.
// clang-23 toolchain: 6-arg builtin (g0, g1, g2, g3, g4, cpol).
// ---------------------------------------------------------------------------
static __device__ __forceinline__ void tdm_load_2d_bf16(
        const void* gptr, unsigned int lds_byte_addr,
        unsigned int tile_d0, unsigned int tile_d1,
        unsigned long long stride0_elems) {
    unsigned long long ga = (unsigned long long)gptr;
    const unsigned int td0 = 0x100000u;   // large tensor dims: no OOB clamp
    const unsigned int td1 = 0x100000u;
    v4u g0;
    g0.x = 1u;                                             // count=1, user desc
    g0.y = lds_byte_addr;                                  // lds_addr
    g0.z = (unsigned int)(ga & 0xFFFFFFFFu);               // global_addr[31:0]
    g0.w = (unsigned int)((ga >> 32) & 0x1FFFFFFu)         // global_addr[56:32]
         | (2u << 30);                                     // type=2 ("image")
    v8i g1;
    g1[0] = (int)(1u << 16);                               // data_size=1 (2B)
    g1[1] = (int)((td0 & 0xFFFFu) << 16);                  // tensor_dim0[15:0]
    g1[2] = (int)((td0 >> 16) | ((td1 & 0xFFFFu) << 16));  // dim0[31:16]|dim1[15:0]
    g1[3] = (int)((td1 >> 16) | (tile_d0 << 16));          // dim1[31:16]|tile_dim0
    g1[4] = (int)(tile_d1 & 0xFFFFu);                      // tile_dim1; tile_dim2=0
    g1[5] = (int)(unsigned int)(stride0_elems & 0xFFFFFFFFull);     // stride0[31:0]
    g1[6] = (int)(unsigned int)((stride0_elems >> 32) & 0xFFFFull); // stride0[47:32]
    g1[7] = 0;                                             // stride1=0 (2D)
    v4i z4 = {0, 0, 0, 0};
    v8i z8 = {0, 0, 0, 0, 0, 0, 0, 0};
    __builtin_amdgcn_tensor_load_to_lds(g0, g1, z4, z4, z8, 0);
}
#endif

// ---------------------------------------------------------------------------
// Kernel 0: f32 -> bf16 convert (grid-stride)
// ---------------------------------------------------------------------------
__global__ void f32_to_bf16_kernel(const float* __restrict__ in,
                                   bf16* __restrict__ out, int n) {
    int i = blockIdx.x * blockDim.x + threadIdx.x;
    int stride = gridDim.x * blockDim.x;
    for (; i < n; i += stride) out[i] = (bf16)in[i];
}

// ---------------------------------------------------------------------------
// Kernel 1/3: C[m][n] = sum_k A[m][k] * W[n][k]   (y = A @ W^T)
// Block = 256 threads = 8 waves. Tile = 128(M) x 64(N), K-chunks of 32.
// Double-buffered LDS staged by TDM (wave 0 issues; TENSORcnt + barrier).
// ---------------------------------------------------------------------------
template <bool OUT_F32>
__global__ __launch_bounds__(256) void gemm_xwT_kernel(
        const bf16* __restrict__ A, const bf16* __restrict__ W,
        void* __restrict__ Cout, int M, int N, int K) {
    __shared__ bf16 As[2][128][32];
    __shared__ bf16 Ws[2][64][32];

    const int tid  = threadIdx.x;
    const int wave = tid >> 5;
    const int lane = tid & 31;
    const int g    = lane >> 4;   // K-half selector (fragment layout)
    const int ln   = lane & 15;
    const int mbase = blockIdx.y * 128;
    const int nbase = blockIdx.x * 64;
    const bool tdmWave = (tid < 32);
    (void)tdmWave;

    v8f acc[4];
#pragma unroll
    for (int n = 0; n < 4; ++n) acc[n] = (v8f){0.f,0.f,0.f,0.f,0.f,0.f,0.f,0.f};

#if HAVE_TDM
    if (tdmWave) {
        tdm_load_2d_bf16(&A[(size_t)mbase * K],
                         (unsigned int)(unsigned long long)(void*)&As[0][0][0],
                         32u, 128u, (unsigned long long)K);
        tdm_load_2d_bf16(&W[(size_t)nbase * K],
                         (unsigned int)(unsigned long long)(void*)&Ws[0][0][0],
                         32u, 64u, (unsigned long long)K);
        __builtin_amdgcn_s_wait_tensorcnt(0);
    }
    __syncthreads();
#endif

    int cur = 0;
    for (int kt = 0; kt < K; kt += 32) {
#if HAVE_TDM
        // Kick off DMA of the NEXT K-chunk into the other buffer
        if (kt + 32 < K && tdmWave) {
            tdm_load_2d_bf16(&A[(size_t)mbase * K + kt + 32],
                             (unsigned int)(unsigned long long)(void*)&As[cur ^ 1][0][0],
                             32u, 128u, (unsigned long long)K);
            tdm_load_2d_bf16(&W[(size_t)nbase * K + kt + 32],
                             (unsigned int)(unsigned long long)(void*)&Ws[cur ^ 1][0][0],
                             32u, 64u, (unsigned long long)K);
        }
#else
        // Fallback: manual staging into buffer 0
#pragma unroll
        for (int i = 0; i < 2; ++i) {
            int idx = tid + i * 256;
            int r = idx >> 2, c = (idx & 3) << 3;
            *(v8bf*)&As[0][r][c] =
                *(const v8bf*)&A[(size_t)(mbase + r) * K + kt + c];
        }
        {
            int r = tid >> 2, c = (tid & 3) << 3;
            *(v8bf*)&Ws[0][r][c] =
                *(const v8bf*)&W[(size_t)(nbase + r) * K + kt + c];
        }
        __syncthreads();
#endif

        // Compute on current buffer
        const int am = wave * 16 + ln;
        v16bf aF = cat8(*(const v8bf*)&As[cur][am][g * 8],
                        *(const v8bf*)&As[cur][am][16 + g * 8]);
#pragma unroll
        for (int n = 0; n < 4; ++n) {
            const int bn = n * 16 + ln;
            v16bf bF = cat8(*(const v8bf*)&Ws[cur][bn][g * 8],
                            *(const v8bf*)&Ws[cur][bn][16 + g * 8]);
            acc[n] = __builtin_amdgcn_wmma_f32_16x16x32_bf16(
                         false, aF, false, bF, (short)0, acc[n], false, false);
        }

#if HAVE_TDM
        if (tdmWave) __builtin_amdgcn_s_wait_tensorcnt(0);
        __syncthreads();
        cur ^= 1;
#else
        __syncthreads();
#endif
    }

    // C layout: VGPR r -> row = r + 8*(lane/16), col = lane%16
#pragma unroll
    for (int n = 0; n < 4; ++n)
#pragma unroll
        for (int r = 0; r < 8; ++r) {
            size_t row = (size_t)(mbase + wave * 16 + r + 8 * g);
            size_t col = (size_t)(nbase + n * 16 + ln);
            if (OUT_F32)
                ((float*)Cout)[row * N + col] = acc[n][r];
            else
                ((bf16*)Cout)[row * N + col] = (bf16)acc[n][r];
        }
}

// ---------------------------------------------------------------------------
// Kernel 2: causal flash attention, one block per (b, h, 64-row query tile).
// 128 threads = 4 waves; wave w owns query rows [qbase+16w, qbase+16w+16).
// Q tile + K chunks staged by TDM; V loaded+transposed manually.
// ---------------------------------------------------------------------------
__global__ __launch_bounds__(128) void attn_kernel(
        const bf16* __restrict__ Q, const bf16* __restrict__ Km,
        const bf16* __restrict__ V, bf16* __restrict__ Ctx) {
    __shared__ bf16 Qs[64][64];
    __shared__ bf16 Ks[64][64];
    __shared__ bf16 VTs[64][64];      // transposed V chunk: VTs[d][k]
    __shared__ bf16 Ps[4][16][64];    // per-wave P slice (A-row-major)

    const int tid  = threadIdx.x;
    const int wave = tid >> 5;        // 0..3
    const int lane = tid & 31;
    const int g    = lane >> 4;
    const int ln   = lane & 15;
    const int qt = blockIdx.x;        // query tile (64 rows)
    const int h  = blockIdx.y;
    const int b  = blockIdx.z;
    const int qbase = qt * 64;
    const size_t base = ((size_t)b * S_) * D_ + (size_t)h * HD_;
    const float scale = 0.125f;       // 1/sqrt(64)
    const bool tdmWave = (tid < 32);
    (void)tdmWave;

#if HAVE_TDM
    if (tdmWave)
        tdm_load_2d_bf16(&Q[base + (size_t)qbase * D_],
                         (unsigned int)(unsigned long long)(void*)&Qs[0][0],
                         64u, 64u, (unsigned long long)D_);
#else
#pragma unroll
    for (int i = 0; i < 4; ++i) {
        int idx = tid + i * 128;
        int r = idx >> 3, c = (idx & 7) << 3;
        *(v8bf*)&Qs[r][c] =
            *(const v8bf*)&Q[base + (size_t)(qbase + r) * D_ + c];
    }
#endif

    float mrow[8], lrow[8];
    v8f o[4];
#pragma unroll
    for (int r = 0; r < 8; ++r) { mrow[r] = -1e30f; lrow[r] = 0.f; }
#pragma unroll
    for (int n = 0; n < 4; ++n) o[n] = (v8f){0.f,0.f,0.f,0.f,0.f,0.f,0.f,0.f};

    for (int kc = 0; kc <= qt; ++kc) {
        const int kcbase = kc * 64;
#if HAVE_TDM
        if (tdmWave)
            tdm_load_2d_bf16(&Km[base + (size_t)kcbase * D_],
                             (unsigned int)(unsigned long long)(void*)&Ks[0][0],
                             64u, 64u, (unsigned long long)D_);
#endif
        // V chunk: load row-major, store transposed into LDS
#pragma unroll
        for (int i = 0; i < 4; ++i) {
            int idx = tid + i * 128;
            int r = idx >> 3, c = (idx & 7) << 3;
#if !HAVE_TDM
            *(v8bf*)&Ks[r][c] =
                *(const v8bf*)&Km[base + (size_t)(kcbase + r) * D_ + c];
#endif
            v8bf vv = *(const v8bf*)&V[base + (size_t)(kcbase + r) * D_ + c];
#pragma unroll
            for (int e = 0; e < 8; ++e) VTs[c + e][r] = vv[e];
            if (kc < qt)    // prefetch next V chunk (global_prefetch_b8)
                __builtin_prefetch(&V[base + (size_t)(kcbase + 64 + r) * D_ + c], 0, 1);
        }
#if HAVE_TDM
        if (tdmWave) __builtin_amdgcn_s_wait_tensorcnt(0);
#endif
        __syncthreads();

        // Scores: S(16x64) = Qw(16x64) @ Kchunk^T ; K-dim = HD = 2 WMMA steps
        v8f s[4];
#pragma unroll
        for (int n = 0; n < 4; ++n) s[n] = (v8f){0.f,0.f,0.f,0.f,0.f,0.f,0.f,0.f};
        v16bf aF[2];
#pragma unroll
        for (int ks = 0; ks < 2; ++ks) {
            const int am = wave * 16 + ln;
            aF[ks] = cat8(*(const v8bf*)&Qs[am][ks * 32 + g * 8],
                          *(const v8bf*)&Qs[am][ks * 32 + 16 + g * 8]);
        }
#pragma unroll
        for (int n = 0; n < 4; ++n)
#pragma unroll
            for (int ks = 0; ks < 2; ++ks) {
                const int bn = n * 16 + ln;
                v16bf bF = cat8(*(const v8bf*)&Ks[bn][ks * 32 + g * 8],
                                *(const v8bf*)&Ks[bn][ks * 32 + 16 + g * 8]);
                s[n] = __builtin_amdgcn_wmma_f32_16x16x32_bf16(
                           false, aF[ks], false, bF, (short)0, s[n], false, false);
            }

        // Online softmax, per C-layout row r (global row = qbase+16w+r+8g)
#pragma unroll
        for (int r = 0; r < 8; ++r) {
            const int rowg = qbase + wave * 16 + r + 8 * g;
            float rowmax = -1e30f;
#pragma unroll
            for (int n = 0; n < 4; ++n) {
                float sv = s[n][r] * scale;
                int colg = kcbase + n * 16 + ln;
                if (colg > rowg) sv = -1e30f;   // causal mask
                s[n][r] = sv;
                rowmax = fmaxf(rowmax, sv);
            }
#pragma unroll
            for (int off = 1; off < 16; off <<= 1)
                rowmax = fmaxf(rowmax, __shfl_xor(rowmax, off, 32));
            float mnew  = fmaxf(mrow[r], rowmax);
            float alpha = __expf(mrow[r] - mnew);
            mrow[r] = mnew;
            float rsum = 0.f;
#pragma unroll
            for (int n = 0; n < 4; ++n) {
                float p = __expf(s[n][r] - mnew);
                rsum += p;
                Ps[wave][r + 8 * g][n * 16 + ln] = (bf16)p;  // C-layout -> A-layout
            }
#pragma unroll
            for (int off = 1; off < 16; off <<= 1)
                rsum += __shfl_xor(rsum, off, 32);
            lrow[r] = lrow[r] * alpha + rsum;
#pragma unroll
            for (int n = 0; n < 4; ++n) o[n][r] *= alpha;
        }

        // Context update: O(16x64) += P(16x64) @ Vchunk(64x64)
        v16bf pF[2];
#pragma unroll
        for (int ks = 0; ks < 2; ++ks)
            pF[ks] = cat8(*(const v8bf*)&Ps[wave][ln][ks * 32 + g * 8],
                          *(const v8bf*)&Ps[wave][ln][ks * 32 + 16 + g * 8]);
#pragma unroll
        for (int n = 0; n < 4; ++n)
#pragma unroll
            for (int ks = 0; ks < 2; ++ks) {
                const int bn = n * 16 + ln;     // output dim index
                v16bf vF = cat8(*(const v8bf*)&VTs[bn][ks * 32 + g * 8],
                                *(const v8bf*)&VTs[bn][ks * 32 + 16 + g * 8]);
                o[n] = __builtin_amdgcn_wmma_f32_16x16x32_bf16(
                           false, pF[ks], false, vF, (short)0, o[n], false, false);
            }
        __syncthreads();
    }

    // Normalize and write context (bf16, [tok, D] layout)
#pragma unroll
    for (int n = 0; n < 4; ++n)
#pragma unroll
        for (int r = 0; r < 8; ++r) {
            float val = o[n][r] / lrow[r];
            size_t row = (size_t)(qbase + wave * 16 + r + 8 * g);
            Ctx[((size_t)b * S_ + row) * D_ + h * HD_ + n * 16 + ln] = (bf16)val;
        }
}

// ---------------------------------------------------------------------------
// Launch
// ---------------------------------------------------------------------------
extern "C" void kernel_launch(void* const* d_in, const int* in_sizes, int n_in,
                              void* d_out, int out_size, void* d_ws, size_t ws_size,
                              hipStream_t stream) {
    const float* x  = (const float*)d_in[0];
    const float* wq = (const float*)d_in[1];
    const float* wk = (const float*)d_in[2];
    const float* wv = (const float*)d_in[3];
    const float* wo = (const float*)d_in[4];
    float* out = (float*)d_out;

    const size_t MB = 1024ull * 1024ull;
    char* ws = (char*)d_ws;
    bf16* x_bf  = (bf16*)(ws + 0 * MB);    // 16 MB
    bf16* wq_bf = (bf16*)(ws + 16 * MB);   //  2 MB
    bf16* wk_bf = (bf16*)(ws + 18 * MB);
    bf16* wv_bf = (bf16*)(ws + 20 * MB);
    bf16* wo_bf = (bf16*)(ws + 22 * MB);
    bf16* q_bf  = (bf16*)(ws + 24 * MB);   // 16 MB
    bf16* k_bf  = (bf16*)(ws + 40 * MB);
    bf16* v_bf  = (bf16*)(ws + 56 * MB);
    bf16* c_bf  = (bf16*)(ws + 72 * MB);   // total 88 MB

    const int nX = NT_ * D_;
    const int nW = D_ * D_;
    f32_to_bf16_kernel<<<2048, 256, 0, stream>>>(x,  x_bf,  nX);
    f32_to_bf16_kernel<<<512,  256, 0, stream>>>(wq, wq_bf, nW);
    f32_to_bf16_kernel<<<512,  256, 0, stream>>>(wk, wk_bf, nW);
    f32_to_bf16_kernel<<<512,  256, 0, stream>>>(wv, wv_bf, nW);
    f32_to_bf16_kernel<<<512,  256, 0, stream>>>(wo, wo_bf, nW);

    dim3 gG(D_ / 64, NT_ / 128);   // (16, 64)
    gemm_xwT_kernel<false><<<gG, 256, 0, stream>>>(x_bf, wq_bf, q_bf, NT_, D_, D_);
    gemm_xwT_kernel<false><<<gG, 256, 0, stream>>>(x_bf, wk_bf, k_bf, NT_, D_, D_);
    gemm_xwT_kernel<false><<<gG, 256, 0, stream>>>(x_bf, wv_bf, v_bf, NT_, D_, D_);

    dim3 gA(S_ / 64, H_, B_);      // (32, 16, 4)
    attn_kernel<<<gA, 128, 0, stream>>>(q_bf, k_bf, v_bf, c_bf);

    gemm_xwT_kernel<true><<<gG, 256, 0, stream>>>(c_bf, wo_bf, out, NT_, D_, D_);
}
